// SSMBlock_30880814858921
// MI455X (gfx1250) — compile-verified, matrix operand order checked by reading, not run on hardware
//
#include <hip/hip_runtime.h>

// ---------------------------------------------------------------------------
// Types
// ---------------------------------------------------------------------------
typedef __bf16       v16bf __attribute__((ext_vector_type(16)));
typedef float        v8f   __attribute__((ext_vector_type(8)));
typedef unsigned int u32x4 __attribute__((ext_vector_type(4)));
typedef unsigned short u16;

union FragBF {
    u32x4 q[2];
    v16bf v;
};

#define D_MODEL 1024
#define D_STATE 16
#define BATCH   4
#define SEQ     2048
#define HID     4096
#define MROWS   (BATCH * SEQ)   // 8192

// Flip to 0 if the assembler rejects the async-to-LDS mnemonic.
#define USE_ASYNC_LDS 1

// bf16 round-to-nearest-even (sufficient for GEMM operands)
__device__ __forceinline__ u16 f32_to_bf16(float f) {
    unsigned int u = __float_as_uint(f);
    unsigned int r = u + 0x7FFFu + ((u >> 16) & 1u);
    return (u16)(r >> 16);
}

// ---------------------------------------------------------------------------
// 16B global -> LDS staging. Async path: GLOBAL_LOAD_ASYNC_TO_LDS_B128
// (no VGPR round-trip, tracked by ASYNCcnt). LDS operand is the per-lane
// wave-relative LDS byte offset = low 32 bits of the generic shared pointer.
// ---------------------------------------------------------------------------
__device__ __forceinline__ void stage_b128(u16* lds_dst, const u16* gsrc) {
#if USE_ASYNC_LDS
    unsigned lds_off = (unsigned)(size_t)(void*)lds_dst;
    asm volatile("global_load_async_to_lds_b128 %0, %1, off"
                 :: "v"(lds_off), "v"(gsrc)
                 : "memory");
#else
    *(u32x4*)lds_dst = *(const u32x4*)gsrc;
#endif
}

__device__ __forceinline__ void stage_fence() {
#if USE_ASYNC_LDS
    asm volatile("s_wait_asynccnt 0x0" ::: "memory");
#endif
}

// ---------------------------------------------------------------------------
// f32 -> bf16 conversion (weights, done once per launch)
// ---------------------------------------------------------------------------
__global__ void cvt_f32_bf16_kernel(const float* __restrict__ in,
                                    u16* __restrict__ out, int n) {
    int i = blockIdx.x * blockDim.x + threadIdx.x;
    int stride = gridDim.x * blockDim.x;
    for (; i < n; i += stride) out[i] = f32_to_bf16(in[i]);
}

// ---------------------------------------------------------------------------
// LayerNorm over last dim (1024). One 256-thread block per row.
// ---------------------------------------------------------------------------
template <bool OUT_BF16>
__global__ __launch_bounds__(256) void ln_kernel(const float* __restrict__ x,
                                                 const float* __restrict__ w,
                                                 const float* __restrict__ b,
                                                 void* __restrict__ out) {
    const int row = blockIdx.x;
    const int tid = threadIdx.x;
    const float* xr = x + (size_t)row * D_MODEL;

    float v[4], s = 0.f, s2 = 0.f;
#pragma unroll
    for (int i = 0; i < 4; ++i) {
        v[i] = xr[tid + i * 256];
        s  += v[i];
        s2 += v[i] * v[i];
    }
    __shared__ float r1[256];
    __shared__ float r2[256];
    r1[tid] = s; r2[tid] = s2;
    __syncthreads();
    for (int off = 128; off > 0; off >>= 1) {
        if (tid < off) { r1[tid] += r1[tid + off]; r2[tid] += r2[tid + off]; }
        __syncthreads();
    }
    const float mean = r1[0] * (1.0f / D_MODEL);
    const float var  = r2[0] * (1.0f / D_MODEL) - mean * mean;
    const float inv  = rsqrtf(var + 1e-5f);

#pragma unroll
    for (int i = 0; i < 4; ++i) {
        const int col = tid + i * 256;
        const float o = (v[i] - mean) * inv * w[col] + b[col];
        if (OUT_BF16) ((u16*)out)[(size_t)row * D_MODEL + col] = f32_to_bf16(o);
        else          ((float*)out)[(size_t)row * D_MODEL + col] = o;
    }
}

// ---------------------------------------------------------------------------
// SSM scan + SiLU. One thread per (batch, channel); 16-state in registers.
// ---------------------------------------------------------------------------
__global__ __launch_bounds__(256) void ssm_scan_kernel(
    const float* __restrict__ xn, const float* __restrict__ A_log,
    const float* __restrict__ Bm, const float* __restrict__ Cm,
    const float* __restrict__ Dv, u16* __restrict__ y_bf16) {
    const int idx = blockIdx.x * blockDim.x + threadIdx.x;   // 0..4095
    const int bi = idx / D_MODEL;
    const int d  = idx % D_MODEL;

    float a[D_STATE], bc[D_STATE], cc[D_STATE], h[D_STATE];
#pragma unroll
    for (int n = 0; n < D_STATE; ++n) {
        const float al = A_log[d * D_STATE + n];
        a[n]  = 1.0f / (1.0f + __expf(-al));     // sigmoid(A_log)
        bc[n] = Bm[d * D_STATE + n];
        cc[n] = Cm[d * D_STATE + n];
        h[n]  = 0.0f;
    }
    const float dvec = Dv[d];

    const float* xp = xn + (size_t)bi * SEQ * D_MODEL + d;
    u16* yp = y_bf16 + (size_t)bi * SEQ * D_MODEL + d;

    for (int t = 0; t < SEQ; ++t) {
        const float xt = xp[(size_t)t * D_MODEL];
        float acc = dvec * xt;
#pragma unroll
        for (int n = 0; n < D_STATE; ++n) {
            h[n] = fmaf(a[n], h[n], bc[n] * xt);
            acc  = fmaf(cc[n], h[n], acc);
        }
        const float y = acc / (1.0f + __expf(-acc));   // SiLU
        yp[(size_t)t * D_MODEL] = f32_to_bf16(y);
    }
}

// ---------------------------------------------------------------------------
// WMMA bf16 GEMM:  Out[m,n] = sum_k A[m,k] * W[n,k]  (+ epilogue)
//   EPI == 0 : out_f32[m,n]  = acc + bias[n] + res_f32[m,n]
//   EPI == 1 : out_bf16[m,n] = gelu_exact(acc + bias[n])
// Block tile 128(M) x 64(N), BK = 32, 8 waves (4M x 2N); each wave owns a
// 32x32 output (4 accumulators, A/B fragments each reused x2).
// Double-buffered LDS, async global->LDS staging, one barrier per K-step.
// ---------------------------------------------------------------------------
#define LDA 40            // padded LDS row stride (80B rows, conflict-free)
#define BMT 128
#define BNT 64
#define ASZ (BMT * LDA)   // u16 elements per A buffer
#define BSZ (BNT * LDA)

template <int EPI>
__global__ __launch_bounds__(256) void gemm_bf16_kernel(
    const u16* __restrict__ A, const u16* __restrict__ W,
    const float* __restrict__ bias, const float* res,
    void* out, int N, int K) {
    __shared__ __align__(16) u16 sA[2 * ASZ];
    __shared__ __align__(16) u16 sB[2 * BSZ];

    const int tid  = threadIdx.x;
    const int lane = tid & 31;
    const int wave = tid >> 5;
    const int half = lane >> 4;      // 0: lanes 0-15, 1: lanes 16-31
    const int l16  = lane & 15;

    const int bm = blockIdx.y * BMT;
    const int bn = blockIdx.x * BNT;
    const int m0 = (wave & 3) * 32;      // wave's 32-row M sub-tile
    const int n0 = (wave >> 2) * 32;     // wave's 32-col N slab

    const int srow   = tid >> 2;         // 0..63
    const int schunk = tid & 3;          // 16B chunk within 32 bf16

    // staging source/dest (row srow and row 64+srow of the A tile; row srow of B)
    const u16* gA0 = A + (size_t)(bm + srow) * K + schunk * 8;
    const u16* gA1 = A + (size_t)(bm + 64 + srow) * K + schunk * 8;
    const u16* gW  = W + (size_t)(bn + srow) * K + schunk * 8;
    u16* lA0 = &sA[srow * LDA + schunk * 8];
    u16* lA1 = &sA[(64 + srow) * LDA + schunk * 8];
    u16* lW  = &sB[srow * LDA + schunk * 8];

    auto stage = [&](int buf, int kt) {
        stage_b128(lA0 + buf * ASZ, gA0 + kt);
        stage_b128(lA1 + buf * ASZ, gA1 + kt);
        stage_b128(lW  + buf * BSZ, gW  + kt);
    };

    v8f acc00 = {}, acc01 = {}, acc10 = {}, acc11 = {};

    stage(0, 0);
    stage_fence();
    __syncthreads();

    for (int kt = 0, it = 0; kt < K; kt += 32, ++it) {
        const int buf = it & 1;
        if (kt + 32 < K) stage(buf ^ 1, kt + 32);        // overlap with WMMAs
        if (kt + 64 < K) {                               // global_prefetch_b8
            __builtin_prefetch((const void*)(gA0 + kt + 64), 0, 1);
            __builtin_prefetch((const void*)(gW  + kt + 64), 0, 1);
        }

        // fragments per documented 16-bit A/B layout:
        // lanes 0-15 take K chunks {0-7,16-23}; lanes 16-31 take {8-15,24-31}
        const u16* cA = &sA[buf * ASZ];
        const u16* cB = &sB[buf * BSZ];
        FragBF fa0, fa1, fb0, fb1;
        const u32x4* ra0 = (const u32x4*)&cA[(m0 + l16) * LDA];
        const u32x4* ra1 = (const u32x4*)&cA[(m0 + 16 + l16) * LDA];
        const u32x4* rb0 = (const u32x4*)&cB[(n0 + l16) * LDA];
        const u32x4* rb1 = (const u32x4*)&cB[(n0 + 16 + l16) * LDA];
        fa0.q[0] = ra0[half]; fa0.q[1] = ra0[half + 2];
        fa1.q[0] = ra1[half]; fa1.q[1] = ra1[half + 2];
        fb0.q[0] = rb0[half]; fb0.q[1] = rb0[half + 2];
        fb1.q[0] = rb1[half]; fb1.q[1] = rb1[half + 2];

        acc00 = __builtin_amdgcn_wmma_f32_16x16x32_bf16(
            false, fa0.v, false, fb0.v, (short)0, acc00, false, false);
        acc01 = __builtin_amdgcn_wmma_f32_16x16x32_bf16(
            false, fa0.v, false, fb1.v, (short)0, acc01, false, false);
        acc10 = __builtin_amdgcn_wmma_f32_16x16x32_bf16(
            false, fa1.v, false, fb0.v, (short)0, acc10, false, false);
        acc11 = __builtin_amdgcn_wmma_f32_16x16x32_bf16(
            false, fa1.v, false, fb1.v, (short)0, acc11, false, false);

        stage_fence();      // async writes to buf^1 complete before next iter
        __syncthreads();
    }

    // ---- epilogue: C/D layout -> M = r + 8*half, N = lane&15 ----
    const int gn0 = bn + n0 + l16;
    const int gn1 = gn0 + 16;
    const float b0 = bias[gn0];
    const float b1 = bias[gn1];
#pragma unroll
    for (int r = 0; r < 8; ++r) {
        const int gm0 = bm + m0 + half * 8 + r;
        const int gm1 = gm0 + 16;
        const size_t i00 = (size_t)gm0 * N + gn0;
        const size_t i01 = (size_t)gm0 * N + gn1;
        const size_t i10 = (size_t)gm1 * N + gn0;
        const size_t i11 = (size_t)gm1 * N + gn1;
        if (EPI == 0) {
            float* o = (float*)out;
            o[i00] = acc00[r] + b0 + res[i00];
            o[i01] = acc01[r] + b1 + res[i01];
            o[i10] = acc10[r] + b0 + res[i10];
            o[i11] = acc11[r] + b1 + res[i11];
        } else {
            u16* o = (u16*)out;
            const float v00 = acc00[r] + b0;
            const float v01 = acc01[r] + b1;
            const float v10 = acc10[r] + b0;
            const float v11 = acc11[r] + b1;
            o[i00] = f32_to_bf16(0.5f * v00 * (1.0f + erff(v00 * 0.70710678118654752f)));
            o[i01] = f32_to_bf16(0.5f * v01 * (1.0f + erff(v01 * 0.70710678118654752f)));
            o[i10] = f32_to_bf16(0.5f * v10 * (1.0f + erff(v10 * 0.70710678118654752f)));
            o[i11] = f32_to_bf16(0.5f * v11 * (1.0f + erff(v11 * 0.70710678118654752f)));
        }
    }
}

// ---------------------------------------------------------------------------
// Host launcher
// ---------------------------------------------------------------------------
extern "C" void kernel_launch(void* const* d_in, const int* in_sizes, int n_in,
                              void* d_out, int out_size, void* d_ws, size_t ws_size,
                              hipStream_t stream) {
    const float* x      = (const float*)d_in[0];
    const float* ln_w   = (const float*)d_in[1];
    const float* ln_b   = (const float*)d_in[2];
    const float* A_log  = (const float*)d_in[3];
    const float* B_mat  = (const float*)d_in[4];
    const float* C_mat  = (const float*)d_in[5];
    const float* D_vec  = (const float*)d_in[6];
    const float* out_w  = (const float*)d_in[7];
    const float* out_b  = (const float*)d_in[8];
    const float* ln2_w  = (const float*)d_in[9];
    const float* ln2_b  = (const float*)d_in[10];
    const float* ffn1_w = (const float*)d_in[11];
    const float* ffn1_b = (const float*)d_in[12];
    const float* ffn2_w = (const float*)d_in[13];
    const float* ffn2_b = (const float*)d_in[14];
    float* outp = (float*)d_out;

    // workspace carve-up (256B aligned)
    char* ws = (char*)d_ws;
    size_t off = 0;
    auto carve = [&](size_t bytes) -> void* {
        off = (off + 255) & ~(size_t)255;
        void* p = ws + off;
        off += bytes;
        return p;
    };
    float* xn   = (float*)carve((size_t)MROWS * D_MODEL * 4);  // LN1 output, f32
    u16* actA   = (u16*)  carve((size_t)MROWS * D_MODEL * 2);  // y_bf16, then xn2_bf16
    u16* hdn    = (u16*)  carve((size_t)MROWS * HID * 2);      // gelu(ffn1) bf16
    u16* w_out  = (u16*)  carve((size_t)D_MODEL * D_MODEL * 2);
    u16* w_ffn1 = (u16*)  carve((size_t)HID * D_MODEL * 2);
    u16* w_ffn2 = (u16*)  carve((size_t)D_MODEL * HID * 2);

    // 1) convert weights to bf16 once
    cvt_f32_bf16_kernel<<<512, 256, 0, stream>>>(out_w,  w_out,  D_MODEL * D_MODEL);
    cvt_f32_bf16_kernel<<<2048, 256, 0, stream>>>(ffn1_w, w_ffn1, HID * D_MODEL);
    cvt_f32_bf16_kernel<<<2048, 256, 0, stream>>>(ffn2_w, w_ffn2, D_MODEL * HID);

    // 2) LN1 : x -> xn (f32)
    ln_kernel<false><<<MROWS, 256, 0, stream>>>(x, ln_w, ln_b, (void*)xn);

    // 3) SSM scan + SiLU : xn -> y (bf16)
    ssm_scan_kernel<<<(BATCH * D_MODEL) / 256, 256, 0, stream>>>(
        xn, A_log, B_mat, C_mat, D_vec, actA);

    // 4) GEMM1: x1 = y @ out_w^T + out_b + x   (f32 into d_out)
    gemm_bf16_kernel<0><<<dim3(D_MODEL / BNT, MROWS / BMT), 256, 0, stream>>>(
        actA, w_out, out_b, x, (void*)outp, D_MODEL, D_MODEL);

    // 5) LN2 : x1 -> xn2 (bf16, reuses y buffer)
    ln_kernel<true><<<MROWS, 256, 0, stream>>>(outp, ln2_w, ln2_b, (void*)actA);

    // 6) GEMM2: hdn = gelu(xn2 @ ffn1_w^T + ffn1_b)  (bf16)
    gemm_bf16_kernel<1><<<dim3(HID / BNT, MROWS / BMT), 256, 0, stream>>>(
        actA, w_ffn1, ffn1_b, (const float*)nullptr, (void*)hdn, HID, D_MODEL);

    // 7) GEMM3: out = x1 + hdn @ ffn2_w^T + ffn2_b   (in-place residual on d_out)
    gemm_bf16_kernel<0><<<dim3(D_MODEL / BNT, MROWS / BMT), 256, 0, stream>>>(
        hdn, w_ffn2, ffn2_b, outp, (void*)outp, D_MODEL, HID);
}
